// PoseNet_26096221291161
// MI455X (gfx1250) — matrix operand order repaired and checked by reading, not compiled, not run on hardware
//
#include <hip/hip_runtime.h>

// ---------------------------------------------------------------------------
// PoseNet forward for MI455X (gfx1250).
//  * All GEMMs -> v_wmma_f32_16x16x32_f16, f32 accumulate.
//  * Interior tiles staged with GLOBAL_LOAD_ASYNC_TO_LDS_B128 (ASYNCcnt),
//    double-buffered LDS, one s_wait_asynccnt + barrier per K-step.
//  * f32 tiles in LDS; f16 conversion at fragment load (ds_load_b128 + cvt).
//  * Edge tiles: clamp+select staging (no exec-mask branches).
//  * Bias+ReLU fused in GEMM epilogue; residual fused into BatchNorm.
// ---------------------------------------------------------------------------

typedef __attribute__((ext_vector_type(16))) _Float16 v16h;
typedef __attribute__((ext_vector_type(8)))  float    v8f;

#define TBM 128
#define TBN 64
#define TBK 32
#define KNN 20

// ---------------------------------------------------------------------------
// Generic batched GEMM:  C[b] = A[b] * op(B[b]) (+bias) (ReLU)
// A: [M,lda] row-major (+ b*sA), B: row-major [K,ldb] or (transB) [N,ldb],
// C: [M,ldc] (+ b*sC).  Tile 128x64x32, 256 threads = 8 waves,
// each wave owns a 32x32 C block (2x2 WMMA fragments).
// ---------------------------------------------------------------------------
__global__ __launch_bounds__(256)
void k_gemm_wmma(const float* __restrict__ A, const float* __restrict__ Bm,
                 const float* __restrict__ bias, float* __restrict__ C,
                 int M, int N, int Kd,
                 long sA, long sB, long sC,
                 int lda, int ldb, int ldc,
                 int transB, int relu)
{
    // double-buffered f32 tiles; B stored transposed: Bs[n][k]
    __shared__ __align__(16) float As32[2][TBM][TBK + 4];   // 2*18 KB
    __shared__ __align__(16) float Bs32[2][TBN][TBK + 4];   // 2*9 KB

    const int tid  = threadIdx.x;
    const int lane = tid & 31;
    const int wid  = tid >> 5;
    const int wm   = wid >> 1;              // 0..3 -> 32-row group
    const int wn   = wid & 1;               // 0..1 -> 32-col group
    const int m0   = blockIdx.y * TBM;
    const int n0   = blockIdx.x * TBN;

    A  += (long)blockIdx.z * sA;
    Bm += (long)blockIdx.z * sB;
    C  += (long)blockIdx.z * sC;

    const bool fullM = (m0 + TBM) <= M;
    const bool fullN = (n0 + TBN) <= N;

    auto stage = [&](int k0, int buf) {
        const bool fullK = (k0 + TBK) <= Kd;
        // ---------------- A tile ----------------
        if (fullM && fullK) {
            // 128x32 floats = 1024 float4's, 4 per thread, async -> LDS
            #pragma unroll
            for (int i = 0; i < 4; ++i) {
                int idx = tid + i * 256;               // 0..1023
                int r = idx >> 3, c4 = (idx & 7) << 2;
                unsigned loff = (unsigned)(unsigned long long)(const void*)&As32[buf][r][c4];
                unsigned long long ga =
                    (unsigned long long)(const void*)&A[(long)(m0 + r) * lda + (k0 + c4)];
                asm volatile("global_load_async_to_lds_b128 %0, %1, off"
                             :: "v"(loff), "v"(ga) : "memory");
            }
        } else {
            // clamp + select: unconditional loads, no exec branches
            #pragma unroll
            for (int i = 0; i < 16; ++i) {
                int idx = tid + i * 256;
                int r = idx >> 5, c = idx & 31;
                int gm = m0 + r, gk = k0 + c;
                int gmc = gm < M ? gm : 0;
                int gkc = gk < Kd ? gk : 0;
                float v = A[(long)gmc * lda + gkc];
                As32[buf][r][c] = (gm < M && gk < Kd) ? v : 0.0f;
            }
        }
        // ---------------- B tile (stored as Bs[n][k]) ----------------
        if (fullN && fullK && transB) {
            // B row-major [N,ldb], contiguous in k -> async b128 straight in
            #pragma unroll
            for (int i = 0; i < 2; ++i) {
                int idx = tid + i * 256;               // 0..511
                int nn = idx >> 3, k4 = (idx & 7) << 2;
                unsigned loff = (unsigned)(unsigned long long)(const void*)&Bs32[buf][nn][k4];
                unsigned long long ga =
                    (unsigned long long)(const void*)&Bm[(long)(n0 + nn) * ldb + (k0 + k4)];
                asm volatile("global_load_async_to_lds_b128 %0, %1, off"
                             :: "v"(loff), "v"(ga) : "memory");
            }
        } else if (fullN && fullK) {
            // B row-major [K,ldb]: vector load along n, scatter-transpose to LDS
            #pragma unroll
            for (int i = 0; i < 2; ++i) {
                int idx = tid + i * 256;               // 0..511
                int r = idx >> 4, c4 = (idx & 15) << 2;   // r = k, c4 = n
                const float4 v = *(const float4*)&Bm[(long)(k0 + r) * ldb + (n0 + c4)];
                Bs32[buf][c4 + 0][r] = v.x;
                Bs32[buf][c4 + 1][r] = v.y;
                Bs32[buf][c4 + 2][r] = v.z;
                Bs32[buf][c4 + 3][r] = v.w;
            }
        } else {
            #pragma unroll
            for (int i = 0; i < 8; ++i) {
                int idx = tid + i * 256;
                int r = idx >> 6, c = idx & 63;        // r = k, c = n
                int gk = k0 + r, gn = n0 + c;
                int gkc = gk < Kd ? gk : 0;
                int gnc = gn < N ? gn : 0;
                float v = transB ? Bm[(long)gnc * ldb + gkc]
                                 : Bm[(long)gkc * ldb + gnc];
                Bs32[buf][c][r] = (gk < Kd && gn < N) ? v : 0.0f;
            }
        }
    };

    v8f acc[2][2] = {};

    stage(0, 0);
    asm volatile("s_wait_asynccnt 0x0" ::: "memory");
    __syncthreads();

    const int nk = (Kd + TBK - 1) / TBK;
    for (int t = 0; t < nk; ++t) {
        const int buf = t & 1;
        if (t + 1 < nk) stage((t + 1) * TBK, buf ^ 1);   // overlap copy w/ math

        v16h af[2], bf[2];
        const int rlo = lane & 15;
        const int kha = (lane >> 4) * 8;     // A: K half base 0 / 8
        const int kb  = (lane >> 4) * 8;     // B: K half base 0 / 8
        #pragma unroll
        for (int u = 0; u < 2; ++u) {
            int r = wm * 32 + u * 16 + rlo;
            #pragma unroll
            for (int h = 0; h < 8; ++h) {
                af[u][h]     = (_Float16)As32[buf][r][kha + h];
                af[u][8 + h] = (_Float16)As32[buf][r][16 + kha + h];
            }
            int cc = wn * 32 + u * 16 + rlo;
            #pragma unroll
            for (int h = 0; h < 8; ++h) {
                bf[u][h]     = (_Float16)Bs32[buf][cc][kb + h];
                bf[u][8 + h] = (_Float16)Bs32[buf][cc][16 + kb + h];
            }
        }
        #pragma unroll
        for (int i = 0; i < 2; ++i)
            #pragma unroll
            for (int j = 0; j < 2; ++j)
                acc[i][j] = __builtin_amdgcn_wmma_f32_16x16x32_f16(
                    false, af[i], false, bf[j], (short)0, acc[i][j], false, false);

        asm volatile("s_wait_asynccnt 0x0" ::: "memory");
        __syncthreads();
    }

    // ---- epilogue: bias + optional ReLU ---------------------------------
    if (fullM && fullN) {
        #pragma unroll
        for (int i = 0; i < 2; ++i) {
            int mrow = m0 + wm * 32 + i * 16 + ((lane >> 4) * 8);
            #pragma unroll
            for (int j = 0; j < 2; ++j) {
                int ncol = n0 + wn * 32 + j * 16 + (lane & 15);
                float bv = bias ? bias[ncol] : 0.0f;
                #pragma unroll
                for (int r = 0; r < 8; ++r) {
                    float v = acc[i][j][r] + bv;
                    if (relu) v = fmaxf(v, 0.0f);
                    C[(long)(mrow + r) * ldc + ncol] = v;
                }
            }
        }
    } else {
        #pragma unroll
        for (int i = 0; i < 2; ++i) {
            int mrow = m0 + wm * 32 + i * 16 + ((lane >> 4) * 8);
            #pragma unroll
            for (int j = 0; j < 2; ++j) {
                int ncol = n0 + wn * 32 + j * 16 + (lane & 15);
                if (ncol < N) {
                    float bv = bias ? bias[ncol] : 0.0f;
                    #pragma unroll
                    for (int r = 0; r < 8; ++r) {
                        int mm = mrow + r;
                        if (mm < M) {
                            float v = acc[i][j][r] + bv;
                            if (relu) v = fmaxf(v, 0.0f);
                            C[(long)mm * ldc + ncol] = v;
                        }
                    }
                }
            }
        }
    }
}

// ---------------------------------------------------------------------------
// Training-mode BatchNorm over axis-1 channels of [D0, C, D2] row-major.
// One block per channel; optional ReLU; optional residual add; may be
// in-place (out == x).
// ---------------------------------------------------------------------------
__global__ __launch_bounds__(256)
void k_bn(const float* __restrict__ x, const float* __restrict__ res,
          float* __restrict__ out, int D0, int C, int D2, int relu)
{
    const int c = blockIdx.x;
    __shared__ float s1[256], s2[256];
    const long n = (long)D0 * D2;
    float sum = 0.f, sq = 0.f;
    for (long i = threadIdx.x; i < n; i += 256) {
        long d0 = i / D2, d2 = i % D2;
        float v = x[(d0 * C + c) * D2 + d2];
        sum += v; sq += v * v;
    }
    s1[threadIdx.x] = sum; s2[threadIdx.x] = sq;
    __syncthreads();
    for (int s = 128; s > 0; s >>= 1) {
        if (threadIdx.x < s) { s1[threadIdx.x] += s1[threadIdx.x + s];
                               s2[threadIdx.x] += s2[threadIdx.x + s]; }
        __syncthreads();
    }
    const float mean = s1[0] / (float)n;
    const float var  = s2[0] / (float)n - mean * mean;
    const float rstd = rsqrtf(var + 1e-5f);
    for (long i = threadIdx.x; i < n; i += 256) {
        long d0 = i / D2, d2 = i % D2;
        long o = (d0 * C + c) * D2 + d2;
        float v = (x[o] - mean) * rstd;
        if (relu) v = fmaxf(v, 0.0f);
        if (res)  v += res[o];
        out[o] = v;
    }
}

// ---- squared row norms --------------------------------------------------
__global__ void k_sqnorm(const float* __restrict__ h, float* __restrict__ sq,
                         int BN, int F)
{
    int t = blockIdx.x * 256 + threadIdx.x;
    if (t >= BN) return;
    const float* p = h + (long)t * F;
    float s = 0.f;
    for (int f = 0; f < F; ++f) s += p[f] * p[f];
    sq[t] = s;
}

// ---- per-row top-20 nearest (from Gram matrix) --------------------------
__global__ void k_topk(const float* __restrict__ G, const float* __restrict__ sq,
                       int* __restrict__ idx, int Bb, int N)
{
    int t = blockIdx.x * 256 + threadIdx.x;
    if (t >= Bb * N) return;
    int b = t / N, n = t % N;
    const float* g = G + ((long)b * N + n) * N;
    const float* s = sq + (long)b * N;
    float sn = s[n];
    float best[KNN]; int bi[KNN];
    for (int j = 0; j < KNN; ++j) { best[j] = 3.4e38f; bi[j] = 0; }
    for (int m = 0; m < N; ++m) {
        float d = sn + s[m] - 2.0f * g[m];
        if (d < best[KNN - 1]) {
            int j = KNN - 1;
            while (j > 0 && best[j - 1] > d) {
                best[j] = best[j - 1]; bi[j] = bi[j - 1]; --j;
            }
            best[j] = d; bi[j] = m;
        }
    }
    for (int j = 0; j < KNN; ++j) idx[(long)t * KNN + j] = bi[j];
}

// ---- gather [center, center - neighbor] ---------------------------------
__global__ void k_group(const float* __restrict__ h, const int* __restrict__ idx,
                        float* __restrict__ out, int Bb, int N, int F)
{
    long t = blockIdx.x * 256L + threadIdx.x;        // over B*N*KNN
    if (t >= (long)Bb * N * KNN) return;
    long sn = t / KNN;
    int s = (int)(sn % N), b = (int)(sn / N);
    const float* hc = h + ((long)b * N + s) * F;
    const float* hn = h + ((long)b * N + idx[t]) * F;
    float* o = out + t * (2L * F);
    for (int f = 0; f < F; ++f) { float c = hc[f]; o[f] = c; o[F + f] = c - hn[f]; }
}

// ---- max over k ---------------------------------------------------------
__global__ void k_maxk(const float* __restrict__ in, float* __restrict__ out,
                       int BN, int F)
{
    long t = blockIdx.x * 256L + threadIdx.x;
    if (t >= (long)BN * F) return;
    long s = t / F; int f = (int)(t % F);
    const float* p = in + (s * KNN) * (long)F + f;
    float m = -3.4e38f;
    for (int j = 0; j < KNN; ++j) m = fmaxf(m, p[(long)j * F]);
    out[t] = m;
}

// ---- attention column stats: max & sum-exp over axis n (stride N) -------
__global__ __launch_bounds__(256)
void k_colstats(const float* __restrict__ QK, float* __restrict__ cmax,
                float* __restrict__ csum, int N)
{
    const int m = blockIdx.x, b = blockIdx.y;
    const float* base = QK + (long)b * N * N + m;
    __shared__ float sm[256];
    float mx = -3.4e38f;
    for (int n = threadIdx.x; n < N; n += 256) mx = fmaxf(mx, base[(long)n * N]);
    sm[threadIdx.x] = mx; __syncthreads();
    for (int s = 128; s > 0; s >>= 1) {
        if (threadIdx.x < s) sm[threadIdx.x] = fmaxf(sm[threadIdx.x], sm[threadIdx.x + s]);
        __syncthreads();
    }
    const float M = sm[0]; __syncthreads();
    float se = 0.f;
    for (int n = threadIdx.x; n < N; n += 256) se += __expf(base[(long)n * N] - M);
    sm[threadIdx.x] = se; __syncthreads();
    for (int s = 128; s > 0; s >>= 1) {
        if (threadIdx.x < s) sm[threadIdx.x] += sm[threadIdx.x + s];
        __syncthreads();
    }
    if (threadIdx.x == 0) { cmax[b * N + m] = M; csum[b * N + m] = sm[0]; }
}

// ---- raw QK row sums over axis m (contiguous) ---------------------------
__global__ __launch_bounds__(256)
void k_rowsum(const float* __restrict__ QK, float* __restrict__ rsum, int N)
{
    const int n = blockIdx.x, b = blockIdx.y;
    const float* base = QK + ((long)b * N + n) * N;
    __shared__ float sm[256];
    float s = 0.f;
    for (int m = threadIdx.x; m < N; m += 256) s += base[m];
    sm[threadIdx.x] = s; __syncthreads();
    for (int st = 128; st > 0; st >>= 1) {
        if (threadIdx.x < st) sm[threadIdx.x] += sm[threadIdx.x + st];
        __syncthreads();
    }
    if (threadIdx.x == 0) rsum[b * N + n] = sm[0];
}

// ---- att = exp(QK - colmax)/colsum / rowsum  (in place) -----------------
__global__ void k_attnorm(float* __restrict__ QK, const float* __restrict__ cmax,
                          const float* __restrict__ csum, const float* __restrict__ rsum,
                          int Bb, int N)
{
    long t = blockIdx.x * 256L + threadIdx.x;
    if (t >= (long)Bb * N * N) return;
    int m = (int)(t % N);
    long bn = t / N;
    int n = (int)(bn % N), b = (int)(bn / N);
    QK[t] = __expf(QK[t] - cmax[b * N + m]) / csum[b * N + m] / rsum[b * N + n];
}

// ---- misc elementwise ---------------------------------------------------
__global__ void k_sub(const float* a, const float* b, float* o, long n)
{
    long t = blockIdx.x * 256L + threadIdx.x;
    if (t < n) o[t] = a[t] - b[t];
}
__global__ void k_transpose(const float* __restrict__ src, float* __restrict__ dst,
                            int Bb, int C, int N)   // [B,C,N] -> [B,N,C]
{
    long t = blockIdx.x * 256L + threadIdx.x;
    if (t >= (long)Bb * C * N) return;
    int n = (int)(t % N);
    long bc = t / N;
    int c = (int)(bc % C), b = (int)(bc / C);
    dst[((long)b * N + n) * C + c] = src[t];
}
__global__ void k_copy_cols(const float* __restrict__ src, float* __restrict__ dst,
                            long R, int Cs, int Cd, int off)
{
    long t = blockIdx.x * 256L + threadIdx.x;
    if (t >= R * Cs) return;
    long r = t / Cs; int c = (int)(t % Cs);
    dst[r * Cd + off + c] = src[t];
}
__global__ void k_bcast(const float* __restrict__ ap, float* __restrict__ dst,
                        int Bb, int N, int C, int ld, int off)
{
    long t = blockIdx.x * 256L + threadIdx.x;
    if (t >= (long)Bb * N * C) return;
    int c = (int)(t % C);
    long bn = t / C;
    int n = (int)(bn % N), b = (int)(bn / N);
    dst[((long)b * N + n) * ld + off + c] = ap[(long)b * C + c];
}
__global__ void k_colmean(const float* __restrict__ x, float* __restrict__ out,
                          int Bb, int N, int C)
{
    int t = blockIdx.x * 256 + threadIdx.x;
    if (t >= Bb * C) return;
    int b = t / C, c = t % C;
    float s = 0.f;
    for (int n = 0; n < N; ++n) s += x[((long)b * N + n) * C + c];
    out[t] = s / (float)N;
}
__global__ void k_select(const float* __restrict__ h4, const int* __restrict__ obj,
                         float* __restrict__ out, int Bb, int N, int d)
{
    long t = blockIdx.x * 256L + threadIdx.x;
    if (t >= (long)Bb * d * N) return;
    int n = (int)(t % N);
    long bd = t / N;
    int j = (int)(bd % d), b = (int)(bd / d);
    out[t] = h4[((long)b * N + n) * (21L * d) + (long)obj[b] * d + j];
}

// ---------------------------------------------------------------------------
extern "C" void kernel_launch(void* const* d_in, const int* in_sizes, int n_in,
                              void* d_out, int out_size, void* d_ws, size_t ws_size,
                              hipStream_t stream)
{
    (void)in_sizes; (void)n_in; (void)out_size; (void)ws_size;
    const int Bb = 16, N = 1024, NBN = Bb * N;        // 16384

    const float* x   = (const float*)d_in[0];         // [B,N,3]
    const float* emb = (const float*)d_in[1];         // [B,32,N]
    const int*   obj = (const int*)d_in[2];           // [B]
    auto P = [&](int i) { return (const float*)d_in[i]; };

    // ---- bump arena over d_ws ------------------------------------------
    char* base = (char*)d_ws;
    size_t off = 0;
    auto allocf = [&](size_t n) -> float* {
        float* p = (float*)(base + off);
        off += ((n * sizeof(float) + 255) / 256) * 256;
        return p;
    };
    auto alloci = [&](size_t n) -> int* {
        int* p = (int*)(base + off);
        off += ((n * sizeof(int) + 255) / 256) * 256;
        return p;
    };

    // persistent buffers
    float* bufA = allocf((size_t)NBN * 64);
    float* bufB = allocf((size_t)NBN * 64);
    float* h64  = allocf((size_t)NBN * 64);
    float* xt   = allocf((size_t)NBN * 128);
    float* sq   = allocf((size_t)NBN);
    int*   idx  = alloci((size_t)NBN * KNN);
    float* cmax = allocf((size_t)NBN);
    float* csum = allocf((size_t)NBN);
    float* rsum = allocf((size_t)NBN);
    float* embT = allocf((size_t)NBN * 32);
    float* e1o  = allocf((size_t)NBN * 64);
    float* e2o  = allocf((size_t)NBN * 128);
    float* pf1  = allocf((size_t)NBN * 192);
    float* pf2  = allocf((size_t)NBN * 256);
    float* y1   = allocf((size_t)NBN * 128);
    float* y2   = allocf((size_t)NBN * 128);
    float* feat = allocf((size_t)NBN * 1472);
    float* ap   = allocf((size_t)Bb * 1024);
    const size_t ck = off;                            // phase checkpoint

    auto ew = [](long n) { return dim3((unsigned)((n + 255) / 256)); };
    auto gemm = [&](const float* A, const float* Bm, const float* bias, float* C,
                    int M, int Nn, int Kd, long sA, long sB, long sC,
                    int lda, int ldb, int ldc, int tb, int relu, int batch) {
        dim3 g((Nn + TBN - 1) / TBN, (M + TBM - 1) / TBM, batch);
        k_gemm_wmma<<<g, 256, 0, stream>>>(A, Bm, bias, C, M, Nn, Kd,
                                           sA, sB, sC, lda, ldb, ldc, tb, relu);
    };
    auto bn = [&](const float* xin, const float* res, float* outp,
                  int D0, int Cc, int D2, int relu) {
        k_bn<<<Cc, 256, 0, stream>>>(xin, res, outp, D0, Cc, D2, relu);
    };

    // ==== InputEmbedding ================================================
    gemm(x, P(3), P(4), bufA, NBN, 64, 3, 0, 0, 0, 3, 64, 64, 0, 0, 1);
    bn(bufA, nullptr, bufA, Bb, N, 64, 1);
    gemm(bufA, P(5), P(6), bufB, NBN, 64, 64, 0, 0, 0, 64, 64, 64, 0, 0, 1);
    bn(bufB, nullptr, bufB, Bb, N, 64, 1);

    // ==== SampGroup blocks ==============================================
    auto samp_group = [&](const float* hin, int F, const float* w1, const float* b1,
                          const float* w2, const float* b2, int Fo, float* outp,
                          float* G, float* grp, float* o1, float* o2) {
        // Gram matrix via batched WMMA GEMM (B = hin^T)
        gemm(hin, hin, nullptr, G, N, N, F,
             (long)N * F, (long)N * F, (long)N * N, F, F, N, 1, 0, Bb);
        k_sqnorm<<<ew(NBN), 256, 0, stream>>>(hin, sq, NBN, F);
        k_topk<<<ew(NBN), 256, 0, stream>>>(G, sq, idx, Bb, N);
        const long BNK = (long)NBN * KNN;
        k_group<<<ew(BNK), 256, 0, stream>>>(hin, idx, grp, Bb, N, F);
        const int F2 = 2 * F;
        gemm(grp, w1, b1, o1, (int)BNK, Fo, F2, 0, 0, 0, F2, Fo, Fo, 0, 0, 1);
        bn(o1, nullptr, o1, Bb, N, KNN * Fo, 1);
        gemm(o1, w2, b2, o2, (int)BNK, Fo, Fo, 0, 0, 0, Fo, Fo, Fo, 0, 0, 1);
        bn(o2, nullptr, o2, Bb, N, KNN * Fo, 1);
        k_maxk<<<ew((long)NBN * Fo), 256, 0, stream>>>(o2, outp, NBN, Fo);
    };
    {
        float* G   = allocf((size_t)Bb * N * N);
        float* grp = allocf((size_t)NBN * KNN * 128);
        float* o1  = allocf((size_t)NBN * KNN * 128);
        float* o2  = allocf((size_t)NBN * KNN * 128);
        samp_group(bufB, 64, P(7),  P(8),  P(9),  P(10),  64, h64, G, grp, o1, o2);
        samp_group(h64,  64, P(11), P(12), P(13), P(14), 128, xt,  G, grp, o1, o2);
        off = ck;                                     // release group scratch
    }

    // ==== Attention-phase scratch =======================================
    float* QK  = allocf((size_t)Bb * N * N);
    float* Q   = allocf((size_t)NBN * 512);
    float* Kb_ = allocf((size_t)NBN * 512);
    float* V   = allocf((size_t)NBN * 512);
    float* z   = allocf((size_t)NBN * 512);
    float* df  = allocf((size_t)NBN * 512);
    float* y3  = allocf((size_t)NBN * 256);
    float* xf5 = allocf((size_t)NBN * 512);
    float* y4  = allocf((size_t)NBN * 512);
    float* xf6 = allocf((size_t)NBN * 1024);
    float* t1  = allocf((size_t)NBN * 640);
    float* t2h = allocf((size_t)NBN * 256);
    float* t3  = allocf((size_t)NBN * 128);
    float* t4  = allocf((size_t)NBN * 84);

    auto attn = [&](const float* xin, int C, int pi, float* yout) {
        gemm(xin, P(pi + 0), P(pi + 1), Q,   NBN, C, C, 0, 0, 0, C, C, C, 0, 0, 1);
        gemm(xin, P(pi + 2), P(pi + 3), Kb_, NBN, C, C, 0, 0, 0, C, C, C, 0, 0, 1);
        gemm(xin, P(pi + 4), P(pi + 5), V,   NBN, C, C, 0, 0, 0, C, C, C, 0, 0, 1);
        gemm(Q, Kb_, nullptr, QK, N, N, C,
             (long)N * C, (long)N * C, (long)N * N, C, C, N, 1, 0, Bb);
        k_colstats<<<dim3(N, Bb), 256, 0, stream>>>(QK, cmax, csum, N);
        k_rowsum  <<<dim3(N, Bb), 256, 0, stream>>>(QK, rsum, N);
        k_attnorm <<<ew((long)Bb * N * N), 256, 0, stream>>>(QK, cmax, csum, rsum, Bb, N);
        gemm(QK, V, nullptr, z, N, C, N,
             (long)N * N, (long)N * C, (long)N * C, N, C, C, 0, 0, Bb);
        k_sub<<<ew((long)NBN * C), 256, 0, stream>>>(z, xin, df, (long)NBN * C);
        gemm(df, P(pi + 6), P(pi + 7), z, NBN, C, C, 0, 0, 0, C, C, C, 0, 0, 1);
        bn(z, xin, yout, Bb, N, C, 1);                // y = relu(bn(t)) + x
    };

    // ==== PoseNetFeat ===================================================
    attn(xt, 128, 23, y1);                                            // sa1
    k_transpose<<<ew((long)Bb * 32 * N), 256, 0, stream>>>(emb, embT, Bb, 32, N);
    gemm(embT, P(15), P(16), e1o, NBN, 64, 32, 0, 0, 0, 32, 64, 64, 0, 1, 1);
    k_copy_cols<<<ew((long)NBN * 128), 256, 0, stream>>>(y1,  pf1, NBN, 128, 192, 0);
    k_copy_cols<<<ew((long)NBN * 64),  256, 0, stream>>>(e1o, pf1, NBN, 64,  192, 128);

    attn(y1, 128, 31, y2);                                            // sa2
    gemm(e1o, P(17), P(18), e2o, NBN, 128, 64, 0, 0, 0, 64, 128, 128, 0, 1, 1);
    k_copy_cols<<<ew((long)NBN * 128), 256, 0, stream>>>(y2,  pf2, NBN, 128, 256, 0);
    k_copy_cols<<<ew((long)NBN * 128), 256, 0, stream>>>(e2o, pf2, NBN, 128, 256, 128);

    attn(pf2, 256, 39, y3);                                           // sa3
    gemm(y3, P(19), P(20), xf5, NBN, 512, 256, 0, 0, 0, 256, 512, 512, 0, 1, 1);
    attn(xf5, 512, 47, y4);                                           // sa4
    gemm(y4, P(21), P(22), xf6, NBN, 1024, 512, 0, 0, 0, 512, 1024, 1024, 0, 1, 1);
    k_colmean<<<ew((long)Bb * 1024), 256, 0, stream>>>(xf6, ap, Bb, N, 1024);

    // feat = concat(pf1, pf2, broadcast(ap))
    k_copy_cols<<<ew((long)NBN * 192), 256, 0, stream>>>(pf1, feat, NBN, 192, 1472, 0);
    k_copy_cols<<<ew((long)NBN * 256), 256, 0, stream>>>(pf2, feat, NBN, 256, 1472, 192);
    k_bcast<<<ew((long)NBN * 1024), 256, 0, stream>>>(ap, feat, Bb, N, 1024, 1472, 448);

    // ==== Regression heads =============================================
    float* out_r = (float*)d_out;
    float* out_t = out_r + (size_t)Bb * 4 * N;
    float* out_c = out_t + (size_t)Bb * 3 * N;
    struct Br { int w1, w2, w3, w4, d; float* o; };
    Br br[3] = { {55, 57, 59, 73, 4, out_r},
                 {61, 63, 65, 75, 3, out_t},
                 {67, 69, 71, 77, 1, out_c} };
    for (int i = 0; i < 3; ++i) {
        gemm(feat, P(br[i].w1), P(br[i].w1 + 1), t1,  NBN, 640,  1472, 0,0,0, 1472, 640, 640, 0, 1, 1);
        gemm(t1,   P(br[i].w2), P(br[i].w2 + 1), t2h, NBN, 256,  640,  0,0,0, 640,  256, 256, 0, 1, 1);
        gemm(t2h,  P(br[i].w3), P(br[i].w3 + 1), t3,  NBN, 128,  256,  0,0,0, 256,  128, 128, 0, 1, 1);
        int od = 21 * br[i].d;
        gemm(t3,   P(br[i].w4), P(br[i].w4 + 1), t4,  NBN, od,   128,  0,0,0, 128,  od,  od,  0, 0, 1);
        k_select<<<ew((long)Bb * br[i].d * N), 256, 0, stream>>>(t4, obj, br[i].o, Bb, N, br[i].d);
    }
}
